// Physics_Attention_25744033972851
// MI455X (gfx1250) — matrix operand order repaired and checked by reading, not compiled, not run on hardware
//
#include <hip/hip_runtime.h>
#include <hip/hip_bf16.h>

// ---------------------------------------------------------------------------
// Physics-Attention pipeline for MI455X (gfx1250, wave32, WMMA f16 16x16x32)
// Heavy GEMMs on v_wmma_f32_16x16x32_f16; intermediates in f16 (pipeline is
// HBM-bound: ~1.4GB @ 23.3TB/s ~ 60us). Weight/activation slabs for the two
// 256-deep GEMMs are moved into LDS by the Tensor Data Mover (TDM), issued by
// wave 0 and synchronized with s_wait_tensorcnt + workgroup barrier.
// ---------------------------------------------------------------------------

typedef _Float16 h16;
typedef __attribute__((ext_vector_type(16))) _Float16 v16h;
typedef __attribute__((ext_vector_type(8)))  _Float16 v8h;
typedef __attribute__((ext_vector_type(2)))  _Float16 v2h;
typedef __attribute__((ext_vector_type(8)))  float    v8f;
typedef __attribute__((ext_vector_type(4)))  unsigned int u32x4;
typedef __attribute__((ext_vector_type(8)))  int      i32x8;
typedef __attribute__((ext_vector_type(4)))  int      i32x4;

#define DIMC   256
#define HEADS  8
#define CH     64
#define MSLICE 64
#define INNER  512
#define BATCH  4
#define NTOK   32768
#define BN     (BATCH * NTOK)   // 131072

__device__ __forceinline__ v8f wmma16(v16h a, v16h b, v8f c) {
  return __builtin_amdgcn_wmma_f32_16x16x32_f16(
      /*neg_a=*/false, a, /*neg_b=*/false, b,
      /*c_mod=*/(short)0, c, /*reuse_a=*/false, /*reuse_b=*/false);
}

__device__ __forceinline__ v16h pack16(v8h lo, v8h hi) {
  v16h r;
#pragma unroll
  for (int i = 0; i < 8; ++i) { r[i] = lo[i]; r[8 + i] = hi[i]; }
  return r;
}

// Tensor Data Mover: 2D tile (f16 elements) from global -> LDS.
// tile_w/row_stride in elements. Issued per-wave (scalar path); caller gates
// to one wave and waits TENSORcnt before the block barrier.
__device__ __forceinline__ void tdm_load_2d_f16(unsigned int lds_addr,
                                                const void* gptr,
                                                unsigned int tile_w,
                                                unsigned int tile_h,
                                                unsigned int row_stride) {
  unsigned long long ga = (unsigned long long)gptr;
  u32x4 g0;
  g0[0] = 1u;                                   // count=1, user descriptor
  g0[1] = lds_addr;                             // LDS byte address
  g0[2] = (unsigned int)(ga & 0xffffffffull);   // global_addr[31:0]
  g0[3] = (unsigned int)((ga >> 32) & 0x01ffffffull) | 0x80000000u; // type=2
  i32x8 g1;
  g1[0] = (int)(1u << 16);                      // data_size = 2 bytes
  g1[1] = (int)((tile_w & 0xffffu) << 16);      // tensor_dim0[15:0]
  g1[2] = (int)((tile_w >> 16) | ((tile_h & 0xffffu) << 16));  // td0 hi | td1 lo
  g1[3] = (int)((tile_h >> 16) | (tile_w << 16));              // td1 hi | tile_dim0
  g1[4] = (int)(tile_h & 0xffffu);              // tile_dim1 (tile_dim2 = 0)
  g1[5] = (int)row_stride;                      // tensor_dim0_stride[31:0]
  g1[6] = 0;
  g1[7] = 0;
  i32x4 z4 = {0, 0, 0, 0};
  i32x8 z8 = {0, 0, 0, 0, 0, 0, 0, 0};
  // 6-arg form (clang-23 / therock-10.0 headers lane)
  __builtin_amdgcn_tensor_load_to_lds(g0, g1, z4, z4, z8, 0);
}

__device__ __forceinline__ unsigned int lds_offset_of(const void* p) {
  // generic pointer to __shared__ memory: low 32 bits are the LDS offset
  return (unsigned int)(unsigned long long)p;
}

// ---------------------------------------------------------------------------
// K0: f32 -> f16 weight conversion
// ---------------------------------------------------------------------------
__global__ void k_f32_to_f16(const float* __restrict__ s, h16* __restrict__ d, int n) {
  int i = blockIdx.x * blockDim.x + threadIdx.x;
  if (i < n) d[i] = (h16)s[i];
}

// ---------------------------------------------------------------------------
// K1: Y[b,h,n,c] = f16( x[b,n,:] @ W^T + bias )   (head-major output)
// grid: (BN/64, OUT/64), block 128 (4 waves). W slab arrives by TDM while
// all threads convert the x tile f32->f16 into LDS.
// ---------------------------------------------------------------------------
__global__ __launch_bounds__(128)
void k_proj(const float* __restrict__ x, const h16* __restrict__ W,
            const float* __restrict__ bias, h16* __restrict__ out) {
  __shared__ h16 xs[64 * 256];
  __shared__ h16 ws[64 * 256];
  const int tid  = threadIdx.x;
  const int tok0 = blockIdx.x * 64;
  const int j0   = blockIdx.y * 64;

  // TDM: W slab (64 rows x 256 halfs, contiguous) -> LDS, overlapped with x conv
  if (tid < 32) {
    tdm_load_2d_f16(lds_offset_of(ws), W + (size_t)j0 * 256, 256, 64, 256);
  }

  // stage x tile (64x256 f32 -> f16), contiguous in global
  {
    const float4* xg = (const float4*)(x + (size_t)tok0 * 256);
#pragma unroll
    for (int i = 0; i < 32; ++i) {
      int e = i * 128 + tid;            // 0..4095 float4 chunks
      float4 v = xg[e];
      h16* p = &xs[(e >> 6) * 256 + (e & 63) * 4];
      p[0] = (h16)v.x; p[1] = (h16)v.y; p[2] = (h16)v.z; p[3] = (h16)v.w;
    }
  }
  if (tid < 32) __builtin_amdgcn_s_wait_tensorcnt(0);
  __syncthreads();

  const int wave = tid >> 5, lane = tid & 31;
  const int lh = lane >> 4, lm = lane & 15;
  const int m0 = wave * 16;
  v8f acc[4] = {};

#pragma unroll
  for (int k0 = 0; k0 < 256; k0 += 32) {
    v16h a = pack16(*(const v8h*)&xs[(m0 + lm) * 256 + k0 + lh * 8],
                    *(const v8h*)&xs[(m0 + lm) * 256 + k0 + 16 + lh * 8]);
#pragma unroll
    for (int t = 0; t < 4; ++t) {
      v16h b = pack16(*(const v8h*)&ws[(t * 16 + lm) * 256 + k0 + lh * 8],
                      *(const v8h*)&ws[(t * 16 + lm) * 256 + k0 + 16 + lh * 8]);
      acc[t] = wmma16(a, b, acc[t]);
    }
  }

  const int bb    = tok0 / NTOK;
  const int nrow0 = tok0 % NTOK;
#pragma unroll
  for (int t = 0; t < 4; ++t) {
    int j = j0 + t * 16 + lm;
    int h = j >> 6, c = j & 63;
    float bj = bias[j];
    h16* obase = out + ((size_t)(bb * HEADS + h) * NTOK) * 64 + c;
#pragma unroll
    for (int r = 0; r < 8; ++r) {
      int n = nrow0 + m0 + r + 8 * lh;
      obase[(size_t)n * 64] = (h16)(acc[t][r] + bj);
    }
  }
}

// ---------------------------------------------------------------------------
// K2: logits = xm @ Wslice^T + bslice, /clip(temp), softmax over M=64,
//     store sw (f16), accumulate colsum[b,h,g] (block-reduced atomics)
// grid: B*H*(N/64), block 128
// ---------------------------------------------------------------------------
__global__ __launch_bounds__(128)
void k_slicemax(const h16* __restrict__ xm, const h16* __restrict__ Wsl,
                const float* __restrict__ bslice, const float* __restrict__ temp,
                h16* __restrict__ sw, float* __restrict__ colsum) {
  __shared__ h16   xs[64 * 64];
  __shared__ h16   wl[64 * 64];
  __shared__ float lg[64][65];
  __shared__ float sums[64];
  const int tid = threadIdx.x;
  const int tilesPer = NTOK / 64;
  const int bh   = blockIdx.x / tilesPer;
  const int tok0 = (blockIdx.x % tilesPer) * 64;
  const h16* xb = xm + ((size_t)bh * NTOK + tok0) * 64;
  {
    const v8h* xg = (const v8h*)xb;    v8h* xl = (v8h*)xs;
    const v8h* wg = (const v8h*)Wsl;   v8h* wd = (v8h*)wl;
#pragma unroll
    for (int i = 0; i < 4; ++i) {
      xl[i * 128 + tid] = xg[i * 128 + tid];
      wd[i * 128 + tid] = wg[i * 128 + tid];
    }
  }
  __syncthreads();

  const int wave = tid >> 5, lane = tid & 31;
  const int lh = lane >> 4, lm = lane & 15;
  const int m0 = wave * 16;
  v8f acc[4] = {};
#pragma unroll
  for (int k0 = 0; k0 < 64; k0 += 32) {
    v16h a = pack16(*(const v8h*)&xs[(m0 + lm) * 64 + k0 + lh * 8],
                    *(const v8h*)&xs[(m0 + lm) * 64 + k0 + 16 + lh * 8]);
#pragma unroll
    for (int t = 0; t < 4; ++t) {
      v16h b = pack16(*(const v8h*)&wl[(t * 16 + lm) * 64 + k0 + lh * 8],
                      *(const v8h*)&wl[(t * 16 + lm) * 64 + k0 + 16 + lh * 8]);
      acc[t] = wmma16(a, b, acc[t]);
    }
  }
  const int hh = bh & (HEADS - 1);
  float tc = fminf(fmaxf(temp[hh], 0.1f), 5.0f);
  float rt = 1.0f / tc;
#pragma unroll
  for (int t = 0; t < 4; ++t) {
    int g = t * 16 + lm;
    float bg = bslice[g];
#pragma unroll
    for (int r = 0; r < 8; ++r) lg[m0 + r + 8 * lh][g] = (acc[t][r] + bg) * rt;
  }
  __syncthreads();

  if (tid < 64) {
    float mx = -1e30f;
#pragma unroll
    for (int g = 0; g < 64; ++g) mx = fmaxf(mx, lg[tid][g]);
    float s = 0.f;
#pragma unroll
    for (int g = 0; g < 64; ++g) { float e = __expf(lg[tid][g] - mx); lg[tid][g] = e; s += e; }
    sums[tid] = s;
    float inv = 1.0f / s;
    h16* swp = sw + ((size_t)bh * NTOK + tok0 + tid) * 64;
#pragma unroll
    for (int g = 0; g < 64; ++g) swp[g] = (h16)(lg[tid][g] * inv);
  }
  __syncthreads();
  if (tid < 64) {
    float a = 0.f;
#pragma unroll
    for (int t = 0; t < 64; ++t) a += lg[t][tid] / sums[t];
    atomicAdd(&colsum[(size_t)bh * 64 + tid], a);
  }
}

// ---------------------------------------------------------------------------
// K3: st[b,h,g,c] += sum_n sw[n,g]*fx[n,c]  (K = N contraction)
// Both operands token-major -> transpose through padded LDS (stride 72 halves
// keeps 16-lane column reads conflict free). Partials via f32 atomics.
// grid: (B*H, N/1024), block 128
// ---------------------------------------------------------------------------
#define K3_TILES 16
__global__ __launch_bounds__(128)
void k_slicetoken(const h16* __restrict__ sw, const h16* __restrict__ fx,
                  float* __restrict__ st_acc) {
  __shared__ h16 swT[64 * 72];
  __shared__ h16 fxT[64 * 72];
  const int bh   = blockIdx.x;
  const int tok0 = blockIdx.y * (64 * K3_TILES);
  const h16* swb = sw + ((size_t)bh * NTOK) * 64;
  const h16* fxb = fx + ((size_t)bh * NTOK) * 64;
  const int tid = threadIdx.x, wave = tid >> 5, lane = tid & 31;
  const int lh = lane >> 4, lm = lane & 15;
  v8f acc[4] = {};

  for (int it = 0; it < K3_TILES; ++it) {
    const int t0 = tok0 + it * 64;
    __builtin_prefetch(swb + (size_t)(t0 + 64) * 64, 0, 1);
    __builtin_prefetch(fxb + (size_t)(t0 + 64) * 64, 0, 1);
    __syncthreads();
    // cooperative transpose: token pairs -> b32 packed stores
#pragma unroll
    for (int q = 0; q < 2; ++q) {
      int task = q * 128 + tid;   // 0..255
      int p  = task & 31;         // token pair index
      int g8 = task >> 5;         // 8-feature chunk
      v8h a0 = *(const v8h*)&swb[(size_t)(t0 + 2 * p)     * 64 + g8 * 8];
      v8h a1 = *(const v8h*)&swb[(size_t)(t0 + 2 * p + 1) * 64 + g8 * 8];
      v8h f0 = *(const v8h*)&fxb[(size_t)(t0 + 2 * p)     * 64 + g8 * 8];
      v8h f1 = *(const v8h*)&fxb[(size_t)(t0 + 2 * p + 1) * 64 + g8 * 8];
#pragma unroll
      for (int i = 0; i < 8; ++i) {
        int row = g8 * 8 + i;
        *(v2h*)&swT[row * 72 + 2 * p] = (v2h){a0[i], a1[i]};
        *(v2h*)&fxT[row * 72 + 2 * p] = (v2h){f0[i], f1[i]};
      }
    }
    __syncthreads();
#pragma unroll
    for (int k0 = 0; k0 < 64; k0 += 32) {
      v16h a = pack16(*(const v8h*)&swT[(wave * 16 + lm) * 72 + k0 + lh * 8],
                      *(const v8h*)&swT[(wave * 16 + lm) * 72 + k0 + 16 + lh * 8]);
#pragma unroll
      for (int t = 0; t < 4; ++t) {
        v16h b = pack16(*(const v8h*)&fxT[(t * 16 + lm) * 72 + k0 + lh * 8],
                        *(const v8h*)&fxT[(t * 16 + lm) * 72 + k0 + 16 + lh * 8]);
        acc[t] = wmma16(a, b, acc[t]);
      }
    }
  }
  float* stb = st_acc + (size_t)bh * 4096;
#pragma unroll
  for (int t = 0; t < 4; ++t) {
    int c = t * 16 + lm;
#pragma unroll
    for (int r = 0; r < 8; ++r) {
      int g = wave * 16 + r + 8 * lh;
      atomicAdd(&stb[g * 64 + c], acc[t][r]);
    }
  }
}

// ---------------------------------------------------------------------------
// K4: tiny per-(b,h) 64x64 attention. Normalize slice_token rows by
// 1/(colsum+eps), q/k/v, softmax(qk^T*scale), attn@v, pre-scale output rows
// by 1/(colsum+eps) for the deslice, store transposed (Bt layout) in f16.
// grid 32 blocks, 64 threads
// ---------------------------------------------------------------------------
__global__ __launch_bounds__(64)
void k_attn(const float* __restrict__ st_acc, const float* __restrict__ colsum,
            const float* __restrict__ Wq, const float* __restrict__ Wk,
            const float* __restrict__ Wv, h16* __restrict__ tokT) {
  __shared__ float st[64][65];
  __shared__ float ks[64][65];
  __shared__ float vs[64][65];
  const int bh = blockIdx.x, g = threadIdx.x;
  const float* stb = st_acc + (size_t)bh * 4096;
  float r = 1.0f / (colsum[(size_t)bh * 64 + g] + 1e-5f);
#pragma unroll
  for (int c = 0; c < 64; ++c) st[g][c] = stb[g * 64 + c] * r;
  __syncthreads();

  float q[64];
#pragma unroll
  for (int c = 0; c < 64; ++c) {
    float aq = 0.f, ak = 0.f, av = 0.f;
#pragma unroll
    for (int e = 0; e < 64; ++e) {
      float s = st[g][e];
      aq += s * Wq[c * 64 + e];
      ak += s * Wk[c * 64 + e];
      av += s * Wv[c * 64 + e];
    }
    q[c] = aq; ks[g][c] = ak; vs[g][c] = av;
  }
  __syncthreads();             // st no longer needed -> reuse as attn buffer

  const float scale = 0.125f;  // Ch^-0.5
  float mx = -1e30f;
#pragma unroll
  for (int m = 0; m < 64; ++m) {
    float s = 0.f;
#pragma unroll
    for (int c = 0; c < 64; ++c) s += q[c] * ks[m][c];
    s *= scale;
    st[g][m] = s;
    mx = fmaxf(mx, s);
  }
  float sum = 0.f;
#pragma unroll
  for (int m = 0; m < 64; ++m) { float e = __expf(st[g][m] - mx); st[g][m] = e; sum += e; }
  float inv = 1.0f / sum;

  h16* tb = tokT + (size_t)bh * 4096;
#pragma unroll
  for (int c = 0; c < 64; ++c) {
    float o = 0.f;
#pragma unroll
    for (int m = 0; m < 64; ++m) o += st[g][m] * vs[m][c];
    tb[c * 64 + g] = (h16)(o * inv * r);   // pre-scale for deslice
  }
}

// ---------------------------------------------------------------------------
// K5: out_x[b,n,h*64+c] = sum_g sw[b,h,n,g] * tokT'[c,g]
// grid: B*H*(N/64), block 128
// ---------------------------------------------------------------------------
__global__ __launch_bounds__(128)
void k_deslice(const h16* __restrict__ sw, const h16* __restrict__ tokT,
               h16* __restrict__ outx) {
  __shared__ h16 sws[64 * 64];
  __shared__ h16 tks[64 * 64];
  const int tid = threadIdx.x;
  const int tilesPer = NTOK / 64;
  const int bh   = blockIdx.x / tilesPer;
  const int tok0 = (blockIdx.x % tilesPer) * 64;
  {
    const v8h* sg = (const v8h*)(sw + ((size_t)bh * NTOK + tok0) * 64);
    const v8h* tg = (const v8h*)(tokT + (size_t)bh * 4096);
    v8h* sl = (v8h*)sws; v8h* tl = (v8h*)tks;
#pragma unroll
    for (int i = 0; i < 4; ++i) {
      sl[i * 128 + tid] = sg[i * 128 + tid];
      tl[i * 128 + tid] = tg[i * 128 + tid];
    }
  }
  __syncthreads();
  const int wave = tid >> 5, lane = tid & 31;
  const int lh = lane >> 4, lm = lane & 15;
  const int m0 = wave * 16;
  v8f acc[4] = {};
#pragma unroll
  for (int k0 = 0; k0 < 64; k0 += 32) {
    v16h a = pack16(*(const v8h*)&sws[(m0 + lm) * 64 + k0 + lh * 8],
                    *(const v8h*)&sws[(m0 + lm) * 64 + k0 + 16 + lh * 8]);
#pragma unroll
    for (int t = 0; t < 4; ++t) {
      v16h b = pack16(*(const v8h*)&tks[(t * 16 + lm) * 64 + k0 + lh * 8],
                      *(const v8h*)&tks[(t * 16 + lm) * 64 + k0 + 16 + lh * 8]);
      acc[t] = wmma16(a, b, acc[t]);
    }
  }
  const int bb = bh >> 3, hh = bh & 7;
#pragma unroll
  for (int t = 0; t < 4; ++t) {
    int c = t * 16 + lm;
#pragma unroll
    for (int r = 0; r < 8; ++r) {
      int n = tok0 + m0 + r + 8 * lh;
      outx[((size_t)(bb * NTOK + n)) * 512 + hh * 64 + c] = (h16)acc[t][r];
    }
  }
}

// ---------------------------------------------------------------------------
// K6: out[b,n,:] = out_x @ Wout^T + bout  (f32 output)
// grid: (BN/64, 4), block 128. K=512 in two 256-wide slabs; both slabs are
// strided 2D tiles (row stride 512 halfs) -> moved by TDM into dense LDS.
// ---------------------------------------------------------------------------
__global__ __launch_bounds__(128)
void k_outproj(const h16* __restrict__ outx, const h16* __restrict__ W,
               const float* __restrict__ bout, float* __restrict__ out) {
  __shared__ h16 xs[64 * 256];
  __shared__ h16 ws[64 * 256];
  const int tid  = threadIdx.x;
  const size_t row0 = (size_t)blockIdx.x * 64;
  const int j0   = blockIdx.y * 64;
  const int wave = tid >> 5, lane = tid & 31;
  const int lh = lane >> 4, lm = lane & 15;
  const int m0 = wave * 16;
  v8f acc[4] = {};

  for (int kslab = 0; kslab < 2; ++kslab) {
    __syncthreads();   // previous slab fully consumed before overwrite
    if (tid < 32) {
      tdm_load_2d_f16(lds_offset_of(xs), outx + row0 * 512 + kslab * 256,
                      256, 64, 512);
      tdm_load_2d_f16(lds_offset_of(ws), W + (size_t)j0 * 512 + kslab * 256,
                      256, 64, 512);
      __builtin_amdgcn_s_wait_tensorcnt(0);
    }
    __syncthreads();
#pragma unroll
    for (int k0 = 0; k0 < 256; k0 += 32) {
      v16h a = pack16(*(const v8h*)&xs[(m0 + lm) * 256 + k0 + lh * 8],
                      *(const v8h*)&xs[(m0 + lm) * 256 + k0 + 16 + lh * 8]);
#pragma unroll
      for (int t = 0; t < 4; ++t) {
        v16h b = pack16(*(const v8h*)&ws[(t * 16 + lm) * 256 + k0 + lh * 8],
                        *(const v8h*)&ws[(t * 16 + lm) * 256 + k0 + 16 + lh * 8]);
        acc[t] = wmma16(a, b, acc[t]);
      }
    }
  }
#pragma unroll
  for (int t = 0; t < 4; ++t) {
    int j = j0 + t * 16 + lm;
    float bj = bout[j];
#pragma unroll
    for (int r = 0; r < 8; ++r) {
      size_t n = row0 + m0 + r + 8 * lh;
      out[n * 256 + j] = acc[t][r] + bj;
    }
  }
}

// ---------------------------------------------------------------------------
extern "C" void kernel_launch(void* const* d_in, const int* in_sizes, int n_in,
                              void* d_out, int out_size, void* d_ws, size_t ws_size,
                              hipStream_t stream) {
  const float* x       = (const float*)d_in[0];
  const float* Wfx     = (const float*)d_in[1];
  const float* bfx     = (const float*)d_in[2];
  const float* Wx      = (const float*)d_in[3];
  const float* bx      = (const float*)d_in[4];
  const float* Wslice  = (const float*)d_in[5];
  const float* bslice  = (const float*)d_in[6];
  const float* temp    = (const float*)d_in[7];
  const float* Wq      = (const float*)d_in[8];
  const float* Wk      = (const float*)d_in[9];
  const float* Wv      = (const float*)d_in[10];
  const float* Wout    = (const float*)d_in[11];
  const float* bout    = (const float*)d_in[12];

  // workspace carve (all offsets 256B aligned)
  char* w = (char*)d_ws;
  size_t o = 0;
  h16*   Wfx16  = (h16*)(w + o);  o += (size_t)INNER * DIMC * 2;   // 262144
  h16*   Wx16   = (h16*)(w + o);  o += (size_t)INNER * DIMC * 2;
  h16*   Wout16 = (h16*)(w + o);  o += (size_t)DIMC * INNER * 2;
  h16*   Wsl16  = (h16*)(w + o);  o += (size_t)MSLICE * CH * 2;    // 8192
  h16*   tokT16 = (h16*)(w + o);  o += (size_t)BATCH * HEADS * 64 * 64 * 2;
  float* colsum = (float*)(w + o); o += (size_t)BATCH * HEADS * 64 * 4;
  float* st_acc = (float*)(w + o); o += (size_t)BATCH * HEADS * 4096 * 4;
  h16*   fx16   = (h16*)(w + o);  o += (size_t)BN * INNER * 2;     // 134MB
  h16*   xm16   = (h16*)(w + o);  o += (size_t)BN * INNER * 2;
  h16*   sw16   = (h16*)(w + o);  o += (size_t)BN * INNER * 2;
  h16*   outx16 = (h16*)(w + o);  o += (size_t)BN * INNER * 2;

  (void)hipMemsetAsync(colsum, 0, (size_t)BATCH * HEADS * 64 * 4, stream);
  (void)hipMemsetAsync(st_acc, 0, (size_t)BATCH * HEADS * 4096 * 4, stream);

  // K0: weight conversion
  k_f32_to_f16<<<(INNER * DIMC + 255) / 256, 256, 0, stream>>>(Wfx, Wfx16, INNER * DIMC);
  k_f32_to_f16<<<(INNER * DIMC + 255) / 256, 256, 0, stream>>>(Wx, Wx16, INNER * DIMC);
  k_f32_to_f16<<<(DIMC * INNER + 255) / 256, 256, 0, stream>>>(Wout, Wout16, DIMC * INNER);
  k_f32_to_f16<<<(MSLICE * CH + 255) / 256, 256, 0, stream>>>(Wslice, Wsl16, MSLICE * CH);

  // K1: projections
  dim3 g1(BN / 64, INNER / 64);
  k_proj<<<g1, 128, 0, stream>>>(x, Wfx16, bfx, fx16);
  k_proj<<<g1, 128, 0, stream>>>(x, Wx16, bx, xm16);

  // K2: slice softmax + colsum
  k_slicemax<<<BATCH * HEADS * (NTOK / 64), 128, 0, stream>>>(
      xm16, Wsl16, bslice, temp, sw16, colsum);

  // K3: slice tokens (K=N contraction)
  dim3 g3(BATCH * HEADS, NTOK / (64 * K3_TILES));
  k_slicetoken<<<g3, 128, 0, stream>>>(sw16, fx16, st_acc);

  // K4: tiny attention
  k_attn<<<BATCH * HEADS, 64, 0, stream>>>(st_acc, colsum, Wq, Wk, Wv, tokT16);

  // K5: deslice
  k_deslice<<<BATCH * HEADS * (NTOK / 64), 128, 0, stream>>>(sw16, tokT16, outx16);

  // K6: output projection
  dim3 g6(BN / 64, DIMC / 64);
  k_outproj<<<g6, 128, 0, stream>>>(outx16, Wout16, bout, (float*)d_out);
}
